// ConceptEmbeddingConceptPred_34540126994728
// MI455X (gfx1250) — matrix operand order repaired and checked by reading, not compiled, not run on hardware
//
#include <hip/hip_runtime.h>

#define B_    4096
#define C_    128
#define D_    128
#define H_    8
#define K_    (C_ * D_)    /* 16384 */
#define NOUT  (C_ + H_)    /* 136   */
#define NPAD  144          /* 9 x 16 */
#define NT    9            /* n-tiles of 16 */
#define KSPLIT 8
#define KCHUNK (K_ / KSPLIT) /* 2048 */
#define MBLK  128          /* rows per workgroup (8 waves x 16) */

typedef __attribute__((ext_vector_type(16))) __bf16 v16bf;
typedef __attribute__((ext_vector_type(8)))  float  v8f;

// Phase 0: out[b][n] = bias[n]  (deterministic accumulation target)
__global__ void bias_init_kernel(const float* __restrict__ b_loo,
                                 const float* __restrict__ b_full,
                                 float* __restrict__ out) {
    int idx = blockIdx.x * blockDim.x + threadIdx.x;
    if (idx >= B_ * NOUT) return;
    int n = idx % NOUT;
    out[idx] = (n < C_) ? b_loo[n] : b_full[n - C_];
}

// Phase 1: build W_all (NPAD x K_) in bf16 inside d_ws.
// Row n<128: probe n's LOO weights with a zero row at concept j==n.
// Rows 128..135: W_full. Rows 136..143: zero padding.
__global__ void embed_weights_kernel(const float* __restrict__ W_loo,
                                     const float* __restrict__ W_full,
                                     __bf16* __restrict__ Wall) {
    int idx = blockIdx.x * blockDim.x + threadIdx.x;
    if (idx >= NPAD * K_) return;
    int n = idx >> 14;          // / K_
    int k = idx & (K_ - 1);
    int j = k >> 7;             // / D_
    int d = k & (D_ - 1);
    float w = 0.0f;
    if (n < C_) {
        if (j != n) {
            int src = j - (j > n ? 1 : 0);
            w = W_loo[((long)n * (C_ - 1) + src) * D_ + d];
        }
    } else if (n < NOUT) {
        w = W_full[((long)(n - C_) * C_ + j) * D_ + d];
    }
    Wall[idx] = (__bf16)w;
}

// Phase 2: out += x(4096,16384) * Wall^T via v_wmma_f32_16x16x32_bf16.
// One wave = one 16-row M tile x all 9 N tiles; grid.y splits K.
__global__ void __launch_bounds__(256)
loo_gemm_kernel(const float* __restrict__ x,
                const __bf16* __restrict__ Wall,
                float* __restrict__ out) {
    const int lane  = threadIdx.x & 31;
    const int wave  = threadIdx.x >> 5;
    const int hi    = lane >> 4;        // 0: lanes 0-15, 1: lanes 16-31
    const int ln    = lane & 15;
    const int mbase = blockIdx.x * MBLK + wave * 16;
    const int kbase = blockIdx.y * KCHUNK;

    // A operand (16-bit 16x32 layout): lane row = ln;
    // lanes 0-15 hold K 0..15, lanes 16-31 hold K 16..31 (matches B).
    const float*  xrow = x    + (long)(mbase + ln) * K_ + kbase + hi * 16;
    // B operand (32x16 layout): lane column = ln; lanes 0-15 K 0..15, lanes 16-31 K 16..31,
    // 16 contiguous bf16 per lane -> natural row-major load from Wall.
    const __bf16* wrow = Wall + (long)ln * K_ + kbase + hi * 16;

    v8f acc[NT];
    const v8f vzero = {0.f, 0.f, 0.f, 0.f, 0.f, 0.f, 0.f, 0.f};
#pragma unroll
    for (int t = 0; t < NT; ++t) acc[t] = vzero;

#pragma unroll 2
    for (int ks = 0; ks < KCHUNK; ks += 32) {
        const float4 a0 = *(const float4*)(xrow + ks);
        const float4 a1 = *(const float4*)(xrow + ks + 4);
        const float4 a2 = *(const float4*)(xrow + ks + 8);
        const float4 a3 = *(const float4*)(xrow + ks + 12);
        v16bf A;
        A[0]  = (__bf16)a0.x; A[1]  = (__bf16)a0.y; A[2]  = (__bf16)a0.z; A[3]  = (__bf16)a0.w;
        A[4]  = (__bf16)a1.x; A[5]  = (__bf16)a1.y; A[6]  = (__bf16)a1.z; A[7]  = (__bf16)a1.w;
        A[8]  = (__bf16)a2.x; A[9]  = (__bf16)a2.y; A[10] = (__bf16)a2.z; A[11] = (__bf16)a2.w;
        A[12] = (__bf16)a3.x; A[13] = (__bf16)a3.y; A[14] = (__bf16)a3.z; A[15] = (__bf16)a3.w;
#pragma unroll
        for (int t = 0; t < NT; ++t) {
            const v16bf Bv = *(const v16bf*)(wrow + (long)t * 16 * K_ + ks);
            acc[t] = __builtin_amdgcn_wmma_f32_16x16x32_bf16(
                false, A, false, Bv, (short)0, acc[t], false, false);
        }
    }

    // Epilogue: C/D layout -> lane ln = column n, VGPR v = row (v + hi*8).
    // Combine K-split partials with agent-scope f32 atomics onto bias-initialized out.
#pragma unroll
    for (int t = 0; t < NT; ++t) {
        int n = t * 16 + ln;
        if (n < NOUT) {
#pragma unroll
            for (int v = 0; v < 8; ++v) {
                int m = mbase + v + hi * 8;
                __hip_atomic_fetch_add(out + (long)m * NOUT + n, acc[t][v],
                                       __ATOMIC_RELAXED, __HIP_MEMORY_SCOPE_AGENT);
            }
        }
    }
}

extern "C" void kernel_launch(void* const* d_in, const int* in_sizes, int n_in,
                              void* d_out, int out_size, void* d_ws, size_t ws_size,
                              hipStream_t stream) {
    const float* x      = (const float*)d_in[0];
    const float* W_loo  = (const float*)d_in[1];
    const float* b_loo  = (const float*)d_in[2];
    const float* W_full = (const float*)d_in[3];
    const float* b_full = (const float*)d_in[4];
    float* out  = (float*)d_out;
    __bf16* Wall = (__bf16*)d_ws;   // needs NPAD*K_*2 = 4.5 MB of scratch

    bias_init_kernel<<<(B_ * NOUT + 255) / 256, 256, 0, stream>>>(b_loo, b_full, out);
    embed_weights_kernel<<<(NPAD * K_) / 256, 256, 0, stream>>>(W_loo, W_full, Wall);

    dim3 grid(B_ / MBLK, KSPLIT);
    loo_gemm_kernel<<<grid, 256, 0, stream>>>(x, Wall, out);
}
